// MAB_39144331935858
// MI455X (gfx1250) — matrix operand order repaired and checked by reading, not compiled, not run on hardware
//
#include <hip/hip_runtime.h>
#include <hip/hip_bf16.h>
#include <math.h>

typedef __attribute__((ext_vector_type(16))) _Float16 v16h;
typedef __attribute__((ext_vector_type(8)))  float    v8f;

#define HW   16384
#define NBAT 8

__device__ __forceinline__ float sigmoidf_(float x){ return 1.f/(1.f+__expf(-x)); }
__device__ __forceinline__ float lreluf_(float x){ return x >= 0.f ? x : 0.2f*x; }

// ---------------------------------------------------------------------------
// Weight packing: fp32 OIHW -> f16 in per-lane WMMA A-fragment order:
//   wp[((chunk*4 + ocTile)*32 + lane)*16 + j]
// so each lane's 16 f16 (32B) are contiguous -> 2x global_load_b128, no LDS.
// j = 2*i+p maps to k = chunk*32 + (lane>>4)*16 + j,
// oc = ocTile*16 + (lane&15)  (contiguous k order, same as B read from LDS).
// ---------------------------------------------------------------------------
__global__ void pack_w_kernel(const float* __restrict__ w, _Float16* __restrict__ wp,
                              int OC, int IC, int KK){
  int Ktot = IC*KK;
  int nchunk = (Ktot + 31) >> 5;
  int total = nchunk * 2048;
  int idx = blockIdx.x*256 + threadIdx.x;
  if (idx >= total) return;
  int j = idx & 15;
  int l = (idx >> 4) & 31;
  int t = (idx >> 9) & 3;
  int q = idx >> 11;
  int i = j >> 1, p = j & 1;
  int half = l >> 4, m = l & 15;
  int kq = half*16 + i*2 + p;
  int k = q*32 + kq;
  int oc = t*16 + m;
  float v = 0.f;
  if (oc < OC && k < Ktot){
    int ic = k / KK, r = k - ic*KK;
    v = w[((size_t)oc*IC + ic)*KK + r];
  }
  wp[idx] = (_Float16)v;
}

// ---------------------------------------------------------------------------
// Sobel stage: channel-sum then 4 stencils (rank-1 structure of sobel weights)
// ---------------------------------------------------------------------------
__global__ void xsum_kernel(const float* __restrict__ x, float* __restrict__ xsum){
  int i = blockIdx.x*256 + threadIdx.x;           // over N*HW
  if (i >= NBAT*HW) return;
  int n = i >> 14, hw = i & (HW-1);
  const float* p = x + (size_t)n*64*HW + hw;
  float s = 0.f;
  #pragma unroll 8
  for (int c = 0; c < 64; ++c) s += p[(size_t)c*HW];
  xsum[i] = s;
}

__global__ void sobel_concat_kernel(const float* __restrict__ x, const float* __restrict__ xsum,
                                    const float* __restrict__ factor, const float* __restrict__ sbias,
                                    float* __restrict__ xs){
  int i = blockIdx.x*256 + threadIdx.x;           // over N*HW
  if (i >= NBAT*HW) return;
  int n = i >> 14, hw = i & (HW-1);
  int h = hw >> 7, w = hw & 127;
  float nb[3][3];
  #pragma unroll
  for (int dh = -1; dh <= 1; ++dh)
    #pragma unroll
    for (int dw = -1; dw <= 1; ++dw){
      int ih = h+dh, iw = w+dw;
      nb[dh+1][dw+1] = ((unsigned)ih < 128u && (unsigned)iw < 128u)
                       ? xsum[(n<<14) + ih*128 + iw] : 0.f;
    }
  float r0 = -nb[0][0]-2.f*nb[0][1]-nb[0][2] + nb[2][0]+2.f*nb[2][1]+nb[2][2];
  float r1 = -nb[0][0]+nb[0][2] -2.f*nb[1][0]+2.f*nb[1][2] -nb[2][0]+nb[2][2];
  float r2 = -2.f*nb[0][0]-nb[0][1] -nb[1][0]+nb[1][2] + nb[2][1]+2.f*nb[2][2];
  float r3 =  nb[0][1]+2.f*nb[0][2] -nb[1][0]+nb[1][2] -2.f*nb[2][0]-nb[2][1];
  float r[4] = {r0, r1, r2, r3};
  float* dst = xs + (size_t)n*128*HW + hw;
  const float* src = x + (size_t)n*64*HW + hw;
  for (int c = 0; c < 64; ++c) dst[(size_t)c*HW] = src[(size_t)c*HW];
  for (int c = 0; c < 64; ++c) dst[(size_t)(64+c)*HW] = factor[c]*r[c & 3] + sbias[c];
}

// ---------------------------------------------------------------------------
// Implicit-GEMM conv via v_wmma_f32_16x16x32_f16.
// Block = 256 thr = 8 waves; block tile = 64 oc x 64 spatial; K chunks of 32.
// A fragments: direct L2-resident global b128 loads (fragment-packed weights).
// B fragments: LDS transposed [spatial][k], 96B row stride -> 2x ds b128/lane.
// ---------------------------------------------------------------------------
__global__ __launch_bounds__(256)
void conv_wmma_kernel(const float* __restrict__ in0, int c0, int c0Stride, int c0Off,
                      const float* __restrict__ in1, int c1, int c1Stride, int c1Off,
                      const _Float16* __restrict__ wpk, const float* __restrict__ bias,
                      float* __restrict__ out, int ocTot, int ocOff,
                      const float* __restrict__ res, int resStride, int resOff,
                      int kdim, int pad, int dil, int numOcTiles, int doLrelu, int accumulate)
{
  __shared__ __attribute__((aligned(32))) _Float16 Bsh[64][48];  // [spatial][k], 96B stride
  const int tid  = threadIdx.x;
  const int lane = tid & 31;
  const int wid  = __builtin_amdgcn_readfirstlane(tid >> 5);     // scalar wave id
  const int wm   = wid >> 1, wn = wid & 1;
  const int stile = blockIdx.x * 64;
  const int KK   = kdim*kdim;
  const int Ktot = (c0 + c1)*KK;
  const int m    = lane & 15, half = lane >> 4;

  // B staging: this thread always fills column sl, rows kk0 + 4*j (j=0..7)
  const int sl  = tid & 63;
  const int kk0 = tid >> 6;
  const int s   = stile + sl;
  const int sn  = s >> 14, shw = s & (HW-1);
  const int sh  = shw >> 7, sw = shw & 127;

  v8f acc0 = {0.f,0.f,0.f,0.f,0.f,0.f,0.f,0.f};
  v8f acc1 = {0.f,0.f,0.f,0.f,0.f,0.f,0.f,0.f};

  for (int kc = 0; kc < Ktot; kc += 32){
    if (kc + 32 < Ktot)
      __builtin_prefetch(wpk + (size_t)(((kc>>5)+1)*4)*512, 0, 1);
    // ---- stage B (fp32 gather -> f16, implicit im2col) ----
    #pragma unroll
    for (int j = 0; j < 8; ++j){
      int kk = kk0 + 4*j;
      int k  = kc + kk;
      float v = 0.f;
      if (k < Ktot){
        int ic = k / KK, r = k - ic*KK;
        int kh = r / kdim, kw = r - kh*kdim;
        int ih = sh - pad + dil*kh;
        int iw = sw - pad + dil*kw;
        if ((unsigned)ih < 128u && (unsigned)iw < 128u){
          const float* src; int cs, cc;
          if (ic < c0){ src = in0; cs = c0Stride; cc = c0Off + ic; }
          else        { src = in1; cs = c1Stride; cc = c1Off + ic - c0; }
          v = src[((size_t)sn*cs + cc)*HW + ih*128 + iw];
        }
      }
      Bsh[sl][kk] = (_Float16)v;
    }
    __syncthreads();
    // ---- WMMA: A direct from global (fragment-packed), B contiguous from LDS ----
    if (wm < numOcTiles){
      v16h a  = *(const v16h*)(wpk + ((size_t)((kc>>5)*4 + wm)*32 + lane)*16);
      v16h b0 = *(const v16h*)(&Bsh[wn*32 +      m][half*16]);
      v16h b1 = *(const v16h*)(&Bsh[wn*32 + 16 + m][half*16]);
      acc0 = __builtin_amdgcn_wmma_f32_16x16x32_f16(false, a, false, b0,
                                                    (short)0, acc0, false, false);
      acc1 = __builtin_amdgcn_wmma_f32_16x16x32_f16(false, a, false, b1,
                                                    (short)0, acc1, false, false);
    }
    __syncthreads();
  }
  // ---- epilogue: bias + residual + accumulate + lrelu, per C/D layout ----
  if (wm < numOcTiles){
    #pragma unroll
    for (int half2 = 0; half2 < 2; ++half2){
      int sg = stile + wn*32 + half2*16 + m;
      int n = sg >> 14, hw = sg & (HW-1);
      const v8f& acc = half2 ? acc1 : acc0;
      #pragma unroll
      for (int i = 0; i < 8; ++i){
        int oc = wm*16 + half*8 + i;
        float v = acc[i];
        if (bias) v += bias[oc];
        if (res)  v += res[((size_t)n*resStride + resOff + oc)*HW + hw];
        size_t oidx = ((size_t)n*ocTot + ocOff + oc)*HW + hw;
        if (accumulate) v += out[oidx];
        if (doLrelu) v = lreluf_(v);
        out[oidx] = v;
      }
    }
  }
}

// ---------------------------------------------------------------------------
// Reductions / elementwise tail
// ---------------------------------------------------------------------------
__global__ void mean_nc_kernel(const float* __restrict__ t, int C, float* __restrict__ m){
  int b = blockIdx.x;                 // N*C blocks
  int n = b / C, c = b % C;
  const float* p = t + ((size_t)n*C + c)*HW;
  float s = 0.f;
  for (int i = threadIdx.x; i < HW; i += 256) s += p[i];
  __shared__ float sd[256];
  sd[threadIdx.x] = s; __syncthreads();
  for (int st = 128; st > 0; st >>= 1){
    if (threadIdx.x < st) sd[threadIdx.x] += sd[threadIdx.x+st];
    __syncthreads();
  }
  if (threadIdx.x == 0) m[b] = sd[0] / (float)HW;
}

__global__ void att_axis_kernel(const float* __restrict__ xcat, const float* __restrict__ m1,
                                float* __restrict__ fac, int axisIsH){
  int b = blockIdx.x;                 // N*128 blocks
  int n = b >> 7, a = b & 127;
  float s = 0.f;
  for (int i = threadIdx.x; i < 256*128; i += 256){
    int c = i >> 7, o = i & 127;
    float sc = 2.f*sigmoidf_(m1[n*256 + c]);
    size_t base = ((size_t)n*256 + c)*HW;
    float v = axisIsH ? xcat[base + a*128 + o] : xcat[base + o*128 + a];
    s += v*sc;
  }
  __shared__ float sd[256];
  sd[threadIdx.x] = s; __syncthreads();
  for (int st = 128; st > 0; st >>= 1){
    if (threadIdx.x < st) sd[threadIdx.x] += sd[threadIdx.x+st];
    __syncthreads();
  }
  if (threadIdx.x == 0) fac[b] = 2.f*sigmoidf_(sd[0] / (256.f*128.f));
}

__global__ void att_apply_kernel(float* __restrict__ xcat, const float* __restrict__ m1,
                                 const float* __restrict__ fh, const float* __restrict__ fw){
  size_t i = (size_t)blockIdx.x*256 + threadIdx.x;   // N*256*HW
  int hw = (int)(i & (HW-1)); size_t t = i >> 14;
  int c = (int)(t & 255), n = (int)(t >> 8);
  float sc = 2.f*sigmoidf_(m1[n*256 + c]);
  int h = hw >> 7, w = hw & 127;
  float v = xcat[i]*sc;
  xcat[i] = v*(1.f + fh[n*128 + h] + fw[n*128 + w]);
}

__global__ void gn_stats_kernel(const float* __restrict__ y, float* __restrict__ gm,
                                float* __restrict__ gs){
  int b = blockIdx.x;                 // N*16 blocks, group = 4 contiguous channels
  int n = b >> 4, g = b & 15;
  const float* p = y + ((size_t)n*64 + g*4)*HW;
  float s = 0.f, q = 0.f;
  for (int i = threadIdx.x; i < 4*HW; i += 256){ float v = p[i]; s += v; q += v*v; }
  __shared__ float sd[256], sq[256];
  sd[threadIdx.x] = s; sq[threadIdx.x] = q; __syncthreads();
  for (int st = 128; st > 0; st >>= 1){
    if (threadIdx.x < st){ sd[threadIdx.x] += sd[threadIdx.x+st]; sq[threadIdx.x] += sq[threadIdx.x+st]; }
    __syncthreads();
  }
  if (threadIdx.x == 0){
    float cnt = 4.f*HW;
    float mean = sd[0]/cnt;
    float var  = (sq[0] - cnt*mean*mean)/(cnt - 1.f);   // ddof=1
    gm[b] = mean; gs[b] = sqrtf(fmaxf(var, 0.f));
  }
}

__global__ void gsum_kernel(const float* __restrict__ gamma, float* __restrict__ gsum){
  __shared__ float sd[64];
  sd[threadIdx.x] = gamma[threadIdx.x]; __syncthreads();
  for (int st = 32; st > 0; st >>= 1){
    if (threadIdx.x < st) sd[threadIdx.x] += sd[threadIdx.x+st];
    __syncthreads();
  }
  if (threadIdx.x == 0) gsum[0] = sd[0];
}

__global__ void sru_shuffle_kernel(const float* __restrict__ y, const float* __restrict__ gm,
                                   const float* __restrict__ gs, const float* __restrict__ gamma,
                                   const float* __restrict__ beta, const float* __restrict__ gsum,
                                   float* __restrict__ y2){
  size_t i = (size_t)blockIdx.x*256 + threadIdx.x;   // N*32*HW
  int hw = (int)(i & (HW-1)); size_t t = i >> 14;
  int c = (int)(t & 31), n = (int)(t >> 5);
  float ginv = 1.f / gsum[0];
  int ca = c, cb = c + 32;
  float ya = y[((size_t)n*64 + ca)*HW + hw];
  float yb = y[((size_t)n*64 + cb)*HW + hw];
  int ga = ca >> 2, gb = cb >> 2;
  float gna = (ya - gm[n*16+ga])/(gs[n*16+ga] + 1e-10f)*gamma[ca] + beta[ca];
  float gnb = (yb - gm[n*16+gb])/(gs[n*16+gb] + 1e-10f)*gamma[cb] + beta[cb];
  bool ia = (gna*gamma[ca]*ginv) >= 0.f;    // sigmoid(z)>=0.5 <=> z>=0
  bool ib = (gnb*gamma[cb]*ginv) >= 0.f;
  float xia = ia ? ya : 0.f, xna = ia ? 0.f : ya;
  float xib = ib ? yb : 0.f, xnb = ib ? 0.f : yb;
  y2[((size_t)n*64 + ca)*HW + hw] = xia + xnb;
  y2[((size_t)n*64 + cb)*HW + hw] = xib + xna;
}

__global__ void copy_ch_kernel(const float* __restrict__ src, int srcC, int srcOff,
                               float* __restrict__ dst, int dstC, int dstOff, int nch){
  size_t i = (size_t)blockIdx.x*256 + threadIdx.x;   // N*nch*HW
  int hw = (int)(i & (HW-1)); size_t t = i >> 14;
  int c = (int)(t % nch), n = (int)(t / nch);
  dst[((size_t)n*dstC + dstOff + c)*HW + hw] = src[((size_t)n*srcC + srcOff + c)*HW + hw];
}

__global__ void softmax128_kernel(const float* __restrict__ cm, float* __restrict__ s){
  int n = blockIdx.x, c = threadIdx.x;
  float v = cm[n*128 + c];
  __shared__ float red[128];
  red[c] = v; __syncthreads();
  for (int st = 64; st > 0; st >>= 1){ if (c < st) red[c] = fmaxf(red[c], red[c+st]); __syncthreads(); }
  float mx = red[0]; __syncthreads();
  float e = __expf(v - mx);
  red[c] = e; __syncthreads();
  for (int st = 64; st > 0; st >>= 1){ if (c < st) red[c] += red[c+st]; __syncthreads(); }
  s[n*128 + c] = e / red[0];
}

__global__ void sm_combine_kernel(const float* __restrict__ outc, const float* __restrict__ s,
                                  float* __restrict__ y3){
  size_t i = (size_t)blockIdx.x*256 + threadIdx.x;   // N*64*HW
  int hw = (int)(i & (HW-1)); size_t t = i >> 14;
  int c = (int)(t & 63), n = (int)(t >> 6);
  float a = s[n*128 + c]      * outc[((size_t)n*128 + c)*HW + hw];
  float b = s[n*128 + c + 64] * outc[((size_t)n*128 + c + 64)*HW + hw];
  y3[i] = a + b;
}

__global__ void bn_stats_kernel(const float* __restrict__ y3, float* __restrict__ bm,
                                float* __restrict__ bv){
  int c = blockIdx.x;                 // 64 blocks
  float s = 0.f, q = 0.f;
  for (int i = threadIdx.x; i < NBAT*HW; i += 256){
    int n = i >> 14, hw = i & (HW-1);
    float v = y3[((size_t)n*64 + c)*HW + hw];
    s += v; q += v*v;
  }
  __shared__ float sd[256], sq[256];
  sd[threadIdx.x] = s; sq[threadIdx.x] = q; __syncthreads();
  for (int st = 128; st > 0; st >>= 1){
    if (threadIdx.x < st){ sd[threadIdx.x] += sd[threadIdx.x+st]; sq[threadIdx.x] += sq[threadIdx.x+st]; }
    __syncthreads();
  }
  if (threadIdx.x == 0){
    float cnt = (float)(NBAT*HW);
    float mean = sd[0]/cnt;
    bm[c] = mean; bv[c] = sq[0]/cnt - mean*mean;       // ddof=0
  }
}

__global__ void bn_final_kernel(const float* __restrict__ y3, const float* __restrict__ bm,
                                const float* __restrict__ bv, const float* __restrict__ gamma,
                                const float* __restrict__ beta, float* __restrict__ out){
  size_t i = (size_t)blockIdx.x*256 + threadIdx.x;   // N*64*HW
  size_t t = i >> 14; int c = (int)(t & 63);
  float v = (y3[i] - bm[c]) * rsqrtf(bv[c] + 1e-5f) * gamma[c] + beta[c];
  out[i] = lreluf_(v);
}

// ---------------------------------------------------------------------------
// Host orchestration
// ---------------------------------------------------------------------------
static inline size_t align256(size_t x){ return (x + 255) & ~(size_t)255; }
static inline size_t packedBytes(int IC, int KK){
  size_t nchunk = ((size_t)IC*KK + 31) >> 5;
  return nchunk * 2048 * 2;
}

extern "C" void kernel_launch(void* const* d_in, const int* in_sizes, int n_in,
                              void* d_out, int out_size, void* d_ws, size_t ws_size,
                              hipStream_t stream)
{
  (void)in_sizes; (void)n_in; (void)out_size; (void)ws_size;
  const float* x            = (const float*)d_in[0];
  const float* sobel_factor = (const float*)d_in[1];
  const float* sobel_bias   = (const float*)d_in[2];
  const float* w_p1  = (const float*)d_in[3];   const float* b_p1  = (const float*)d_in[4];
  const float* w_f1  = (const float*)d_in[5];   const float* b_f1  = (const float*)d_in[6];
  const float* w_c1  = (const float*)d_in[7];   const float* b_c1  = (const float*)d_in[8];
  const float* w_f2  = (const float*)d_in[9];   const float* b_f2  = (const float*)d_in[10];
  const float* w_c2  = (const float*)d_in[11];  const float* b_c2  = (const float*)d_in[12];
  const float* w_f3  = (const float*)d_in[13];  const float* b_f3  = (const float*)d_in[14];
  const float* w_c3  = (const float*)d_in[15];  const float* b_c3  = (const float*)d_in[16];
  const float* w_fus = (const float*)d_in[17];  const float* b_fus = (const float*)d_in[18];
  const float* sru_gamma = (const float*)d_in[19];
  const float* sru_beta  = (const float*)d_in[20];
  const float* w_sq1 = (const float*)d_in[21];
  const float* w_sq2 = (const float*)d_in[22];
  const float* w_gwc = (const float*)d_in[23];  const float* b_gwc = (const float*)d_in[24];
  const float* w_pwc1 = (const float*)d_in[25];
  const float* w_pwc2 = (const float*)d_in[26];
  const float* bn_gamma = (const float*)d_in[27];
  const float* bn_beta  = (const float*)d_in[28];

  char* ws = (char*)d_ws;
  size_t off = 0;
  auto alloc = [&](size_t bytes)->char*{ char* p = ws + off; off = align256(off + bytes); return p; };

  float* xs    = (float*)alloc((size_t)NBAT*128*HW*4);
  float* xcat  = (float*)alloc((size_t)NBAT*256*HW*4);
  float* tmpA  = (float*)alloc((size_t)NBAT*64*HW*4);
  float* yb    = (float*)alloc((size_t)NBAT*64*HW*4);
  float* y2    = (float*)alloc((size_t)NBAT*64*HW*4);
  float* up16  = (float*)alloc((size_t)NBAT*16*HW*4);
  float* low16 = (float*)alloc((size_t)NBAT*16*HW*4);
  float* outc  = (float*)alloc((size_t)NBAT*128*HW*4);
  float* y3    = (float*)alloc((size_t)NBAT*64*HW*4);
  float* xsumb = (float*)alloc((size_t)NBAT*HW*4);
  float* m1    = (float*)alloc(NBAT*256*4);
  float* fh    = (float*)alloc(NBAT*128*4);
  float* fw    = (float*)alloc(NBAT*128*4);
  float* gm    = (float*)alloc(NBAT*16*4);
  float* gsd   = (float*)alloc(NBAT*16*4);
  float* gsumv = (float*)alloc(256);
  float* cm    = (float*)alloc(NBAT*128*4);
  float* ssoft = (float*)alloc(NBAT*128*4);
  float* bm    = (float*)alloc(64*4);
  float* bv    = (float*)alloc(64*4);

  _Float16* wp_p1  = (_Float16*)alloc(packedBytes(128,1));
  _Float16* wp_f1  = (_Float16*)alloc(packedBytes(192,9));
  _Float16* wp_f2  = (_Float16*)alloc(packedBytes(192,9));
  _Float16* wp_f3  = (_Float16*)alloc(packedBytes(192,9));
  _Float16* wp_c1  = (_Float16*)alloc(packedBytes(64,9));
  _Float16* wp_c2  = (_Float16*)alloc(packedBytes(64,9));
  _Float16* wp_c3  = (_Float16*)alloc(packedBytes(64,9));
  _Float16* wp_fus = (_Float16*)alloc(packedBytes(256,1));
  _Float16* wp_sq1 = (_Float16*)alloc(packedBytes(32,1));
  _Float16* wp_sq2 = (_Float16*)alloc(packedBytes(32,1));
  _Float16* wp_gw0 = (_Float16*)alloc(packedBytes(8,9));
  _Float16* wp_gw1 = (_Float16*)alloc(packedBytes(8,9));
  _Float16* wp_pw1 = (_Float16*)alloc(packedBytes(16,1));
  _Float16* wp_pw2 = (_Float16*)alloc(packedBytes(16,1));

  auto pack = [&](const float* w, _Float16* wp, int OC, int IC, int KK){
    int tot = (int)(packedBytes(IC,KK)/2);
    pack_w_kernel<<<(tot+255)/256, 256, 0, stream>>>(w, wp, OC, IC, KK);
  };
  pack(w_p1, wp_p1, 64, 128, 1);
  pack(w_f1, wp_f1, 64, 192, 9);  pack(w_f2, wp_f2, 64, 192, 9);  pack(w_f3, wp_f3, 64, 192, 9);
  pack(w_c1, wp_c1, 64, 64, 9);   pack(w_c2, wp_c2, 64, 64, 9);   pack(w_c3, wp_c3, 64, 64, 9);
  pack(w_fus, wp_fus, 64, 256, 1);
  pack(w_sq1, wp_sq1, 16, 32, 1); pack(w_sq2, wp_sq2, 16, 32, 1);
  pack(w_gwc,            wp_gw0, 32, 8, 9);
  pack(w_gwc + 32*8*9,   wp_gw1, 32, 8, 9);
  pack(w_pwc1, wp_pw1, 64, 16, 1);
  pack(w_pwc2, wp_pw2, 48, 16, 1);

  // Sobel + concat -> xs (128 ch)
  xsum_kernel<<<512, 256, 0, stream>>>(x, xsumb);
  sobel_concat_kernel<<<512, 256, 0, stream>>>(x, xsumb, sobel_factor, sobel_bias, xs);

  const int CB = (NBAT*HW) / 64;   // 2048 conv blocks (64-wide spatial tiles)
  #define CONV(...) conv_wmma_kernel<<<CB, 256, 0, stream>>>(__VA_ARGS__)

  // x0 = lrelu(p1(xs))                                -> xcat[0:64]
  CONV(xs,128,128,0,  nullptr,0,0,0,  wp_p1, b_p1,  xcat,256,0,   nullptr,0,0, 1,0,1, 4, 1, 0);
  // out1 = lrelu(f1([xs,x0]))                         -> tmpA
  CONV(xs,128,128,0,  xcat,64,256,0,  wp_f1, b_f1,  tmpA,64,0,    nullptr,0,0, 3,1,1, 4, 1, 0);
  // x1 = lrelu(c1(out1))                              -> xcat[64:128]
  CONV(tmpA,64,64,0,  nullptr,0,0,0,  wp_c1, b_c1,  xcat,256,64,  nullptr,0,0, 3,1,1, 4, 1, 0);
  // out2 = lrelu(f2([xs,x1]))                         -> tmpA
  CONV(xs,128,128,0,  xcat,64,256,64, wp_f2, b_f2,  tmpA,64,0,    nullptr,0,0, 3,1,1, 4, 1, 0);
  // x2 = lrelu(c2(out2)) dil=3                        -> xcat[128:192]
  CONV(tmpA,64,64,0,  nullptr,0,0,0,  wp_c2, b_c2,  xcat,256,128, nullptr,0,0, 3,3,3, 4, 1, 0);
  // out3 = lrelu(f3([xs,x2]))                         -> tmpA
  CONV(xs,128,128,0,  xcat,64,256,128,wp_f3, b_f3,  tmpA,64,0,    nullptr,0,0, 3,1,1, 4, 1, 0);
  // x3 = lrelu(c3(out3)) dil=5                        -> xcat[192:256]
  CONV(tmpA,64,64,0,  nullptr,0,0,0,  wp_c3, b_c3,  xcat,256,192, nullptr,0,0, 3,5,5, 4, 1, 0);

  // Attention on xcat (in place)
  mean_nc_kernel<<<NBAT*256, 256, 0, stream>>>(xcat, 256, m1);
  att_axis_kernel<<<NBAT*128, 256, 0, stream>>>(xcat, m1, fh, 1);  // per (n,h)
  att_axis_kernel<<<NBAT*128, 256, 0, stream>>>(xcat, m1, fw, 0);  // per (n,w)
  att_apply_kernel<<<(NBAT*256*HW)/256, 256, 0, stream>>>(xcat, m1, fh, fw);

  // y = x + fusion(att)
  CONV(xcat,256,256,0, nullptr,0,0,0, wp_fus, b_fus, yb,64,0,     x,64,0,     1,0,1, 4, 0, 0);

  // GroupNorm(16 groups, ddof=1) + SRU gate + channel shuffle -> y2
  gn_stats_kernel<<<NBAT*16, 256, 0, stream>>>(yb, gm, gsd);
  gsum_kernel<<<1, 64, 0, stream>>>(sru_gamma, gsumv);
  sru_shuffle_kernel<<<(NBAT*32*HW)/256, 256, 0, stream>>>(yb, gm, gsd, sru_gamma, sru_beta, gsumv, y2);

  // squeeze convs: up = sq1(y2[:,0:32]) -> 16ch, low = sq2(y2[:,32:64]) -> 16ch
  CONV(y2,32,64,0,   nullptr,0,0,0, wp_sq1, nullptr, up16,16,0,   nullptr,0,0, 1,0,1, 1, 0, 0);
  CONV(y2,32,64,32,  nullptr,0,0,0, wp_sq2, nullptr, low16,16,0,  nullptr,0,0, 1,0,1, 1, 0, 0);

  // Y1 = gwc(up, groups=2) + pwc1(up)   -> outc[0:64]
  CONV(up16,8,16,0,  nullptr,0,0,0, wp_gw0, b_gwc,      outc,128,0,  nullptr,0,0, 3,1,1, 2, 0, 0);
  CONV(up16,8,16,8,  nullptr,0,0,0, wp_gw1, b_gwc + 32, outc,128,32, nullptr,0,0, 3,1,1, 2, 0, 0);
  CONV(up16,16,16,0, nullptr,0,0,0, wp_pw1, nullptr,    outc,128,0,  nullptr,0,0, 1,0,1, 4, 0, 1);
  // Y2 = [pwc2(low), low]               -> outc[64:128]
  CONV(low16,16,16,0,nullptr,0,0,0, wp_pw2, nullptr,    outc,128,64, nullptr,0,0, 1,0,1, 3, 0, 0);
  copy_ch_kernel<<<(NBAT*16*HW)/256, 256, 0, stream>>>(low16,16,0, outc,128,112, 16);

  // channel softmax scaling then fold halves
  mean_nc_kernel<<<NBAT*128, 256, 0, stream>>>(outc, 128, cm);
  softmax128_kernel<<<NBAT, 128, 0, stream>>>(cm, ssoft);
  sm_combine_kernel<<<(NBAT*64*HW)/256, 256, 0, stream>>>(outc, ssoft, y3);

  // BatchNorm over (N,H,W) + lrelu -> d_out
  bn_stats_kernel<<<64, 256, 0, stream>>>(y3, bm, bv);
  bn_final_kernel<<<(NBAT*64*HW)/256, 256, 0, stream>>>(y3, bm, bv, bn_gamma, bn_beta, (float*)d_out);
  #undef CONV
}
